// TriangleAttentionStartingNode_1133871366385
// MI455X (gfx1250) — compile-verified, matrix operand order checked by reading, not run on hardware
//
#include <hip/hip_runtime.h>

typedef __bf16 bf16_t;
typedef __attribute__((ext_vector_type(16))) __bf16 v16bf;
typedef __attribute__((ext_vector_type(8)))  float  v8f;

#define NSEQ 256
#define CDIM 128
#define NH   4
#define DH   32
#define NROWS (NSEQ*NSEQ)
#define WSTR 136   // LDS stride (elems) for 128-wide weight rows: 272B = 16*17 (aligned, conflict-free)
#define KSTR 40    // LDS stride for 32-wide rows: 80B = 16*5
#define VSTR 264   // LDS stride for 256-wide vT rows: 528B = 16*33

union FragBF { v16bf v; int4 q[2]; };
union B4 { bf16_t h[4]; int2 i2; };

__device__ __forceinline__ float fast_rcp(float x) { return __builtin_amdgcn_rcpf(x); }

__device__ __forceinline__ float redmax16(float v){
#pragma unroll
  for (int off=1; off<16; off<<=1) v = fmaxf(v, __shfl_xor(v, off, 32));
  return v;
}
__device__ __forceinline__ float redsum16(float v){
#pragma unroll
  for (int off=1; off<16; off<<=1) v += __shfl_xor(v, off, 32);
  return v;
}
__device__ __forceinline__ float redsum32(float v){
#pragma unroll
  for (int off=1; off<32; off<<=1) v += __shfl_xor(v, off, 32);
  return v;
}

// ---------------- Kernel 1: LayerNorm -> zn (bf16) ----------------
__global__ __launch_bounds__(256)
void ln_kernel(const float* __restrict__ z, const float* __restrict__ w,
               const float* __restrict__ b, bf16_t* __restrict__ zn)
{
  const int wave = threadIdx.x >> 5;
  const int lane = threadIdx.x & 31;
  const size_t row = (size_t)blockIdx.x * 8 + wave;
  const float4 x = *(const float4*)(z + row*CDIM + lane*4);
  float s = x.x + x.y + x.z + x.w;
  s = redsum32(s);
  const float mu = s * (1.0f/CDIM);
  const float dx0 = x.x-mu, dx1 = x.y-mu, dx2 = x.z-mu, dx3 = x.w-mu;
  float ss = dx0*dx0 + dx1*dx1 + dx2*dx2 + dx3*dx3;
  ss = redsum32(ss);
  const float inv = rsqrtf(ss * (1.0f/CDIM) + 1e-5f);
  const float4 wv = *(const float4*)(w + lane*4);
  const float4 bv = *(const float4*)(b + lane*4);
  B4 o;
  o.h[0] = (bf16_t)(dx0*inv*wv.x + bv.x);
  o.h[1] = (bf16_t)(dx1*inv*wv.y + bv.y);
  o.h[2] = (bf16_t)(dx2*inv*wv.z + bv.z);
  o.h[3] = (bf16_t)(dx3*inv*wv.w + bv.w);
  *(int2*)(zn + row*CDIM + lane*4) = o.i2;
}

// ---------------- Kernel 2: fused projections (q,k,vT,g,bias) ----------------
__global__ __launch_bounds__(256)
void proj_kernel(const bf16_t* __restrict__ zn,
                 const float* __restrict__ Wq, const float* __restrict__ Wk,
                 const float* __restrict__ Wv, const float* __restrict__ Wg,
                 const float* __restrict__ Wb, const float* __restrict__ bg,
                 bf16_t* __restrict__ qb, bf16_t* __restrict__ kb,
                 bf16_t* __restrict__ vT, bf16_t* __restrict__ gb,
                 float* __restrict__ biasT)
{
  __shared__ __align__(16) bf16_t wL[CDIM*WSTR];
  __shared__ __align__(16) bf16_t wbL[16*WSTR];
  __shared__ float bgL[CDIM];

  const int tid  = threadIdx.x;
  const int wave = tid >> 5;
  const int lane = tid & 31;
  const int m  = lane & 15;
  const int hf = lane >> 4;
  const int row0 = blockIdx.x * 128 + wave*16;

  // A fragments (zn rows, K=128 -> 4 fragments), kept in registers throughout
  FragBF a[4];
  {
    const bf16_t* p = zn + (size_t)(row0 + m)*CDIM + hf*8;
#pragma unroll
    for (int kk=0; kk<4; ++kk) {
      a[kk].q[0] = *(const int4*)(p + kk*32);
      a[kk].q[1] = *(const int4*)(p + kk*32 + 16);
    }
  }

  for (int idx = tid; idx < CDIM; idx += 256) bgL[idx] = bg[idx];
  for (int idx = tid; idx < 16*CDIM; idx += 256) {
    const int n = idx >> 7, k = idx & 127;
    wbL[n*WSTR + k] = (n < NH) ? (bf16_t)Wb[n*CDIM + k] : (bf16_t)0.0f;
  }

  const float* Ws[4] = {Wq, Wk, Wv, Wg};
#pragma unroll 1
  for (int wsel = 0; wsel < 4; ++wsel) {
    __syncthreads();
    const float* W = Ws[wsel];
    for (int idx = tid; idx < CDIM*CDIM; idx += 256) {
      const int n = idx >> 7, k = idx & 127;
      wL[n*WSTR + k] = (bf16_t)W[idx];
    }
    __syncthreads();

#pragma unroll 1
    for (int nt = 0; nt < 8; ++nt) {
      // preload all 4 B fragments so LDS latency overlaps the WMMA chain
      FragBF bfr[4];
#pragma unroll
      for (int kk = 0; kk < 4; ++kk) {
        const bf16_t* p = wL + (nt*16 + m)*WSTR + kk*32 + hf*8;
        bfr[kk].q[0] = *(const int4*)p;
        bfr[kk].q[1] = *(const int4*)(p + 16);
      }
      v8f acc = {0.f,0.f,0.f,0.f,0.f,0.f,0.f,0.f};
#pragma unroll
      for (int kk = 0; kk < 4; ++kk)
        acc = __builtin_amdgcn_wmma_f32_16x16x32_bf16(false, a[kk].v, false, bfr[kk].v,
                                                      (short)0, acc, false, false);
      const int col = nt*16 + m;
#pragma unroll
      for (int r = 0; r < 8; ++r) {
        const int grow = row0 + r + 8*hf;
        const float x = acc[r];
        if (wsel == 0)      qb[(size_t)grow*CDIM + col] = (bf16_t)x;
        else if (wsel == 1) kb[(size_t)grow*CDIM + col] = (bf16_t)x;
        else if (wsel == 2) {
          const int i = grow >> 8, jc = grow & 255;
          const int h = col >> 5, d = col & 31;
          vT[(((size_t)i*NH + h)*DH + d)*NSEQ + jc] = (bf16_t)x;
        } else {
          // fast sigmoid: v_exp + v_rcp (avoid the precise-div expansion)
          const float g = fast_rcp(1.0f + __expf(-(x + bgL[col])));
          gb[(size_t)grow*CDIM + col] = (bf16_t)g;
        }
      }
    }
  }

  // pair bias: zn @ Wb^T  (one 16-wide N tile, rows>=4 of wbL are zero)
  {
    FragBF bfr[4];
#pragma unroll
    for (int kk = 0; kk < 4; ++kk) {
      const bf16_t* p = wbL + m*WSTR + kk*32 + hf*8;
      bfr[kk].q[0] = *(const int4*)p;
      bfr[kk].q[1] = *(const int4*)(p + 16);
    }
    v8f acc = {0.f,0.f,0.f,0.f,0.f,0.f,0.f,0.f};
#pragma unroll
    for (int kk = 0; kk < 4; ++kk)
      acc = __builtin_amdgcn_wmma_f32_16x16x32_bf16(false, a[kk].v, false, bfr[kk].v,
                                                    (short)0, acc, false, false);
    if (m < NH) {
#pragma unroll
      for (int r = 0; r < 8; ++r)
        biasT[(size_t)m*NROWS + row0 + r + 8*hf] = acc[r];
    }
  }
}

// ---------------- Kernel 3: flash attention per (i, head, j-block) ----------------
__global__ __launch_bounds__(128)
void attn_kernel(const bf16_t* __restrict__ qb, const bf16_t* __restrict__ kb,
                 const bf16_t* __restrict__ vT, const bf16_t* __restrict__ gb,
                 const float* __restrict__ biasT, const float* __restrict__ pm,
                 bf16_t* __restrict__ ob)
{
  __shared__ __align__(16) bf16_t kL[NSEQ*KSTR];
  __shared__ __align__(16) bf16_t vL[DH*VSTR];
  __shared__ __align__(16) bf16_t pL[4*16*KSTR];

  const int tid  = threadIdx.x;
  const int wave = tid >> 5;
  const int lane = tid & 31;
  const int m  = lane & 15;
  const int hf = lane >> 4;
  const int bid = blockIdx.x;
  const int i  = bid >> 4;
  const int h  = (bid >> 2) & 3;
  const int jb = bid & 3;
  const int j0 = jb*64 + wave*16;

  // stage K (256x32) and vT (32x256) for this (i,h)
  for (int c = tid; c < NSEQ*4; c += 128) {        // 16B chunks
    const int row = c >> 2, db = (c & 3)*8;
    *(int4*)&kL[row*KSTR + db] = *(const int4*)&kb[((size_t)(i*NSEQ + row))*CDIM + h*DH + db];
  }
  for (int c = tid; c < DH*32; c += 128) {         // 16B chunks
    const int d = c >> 5, kb8 = (c & 31)*8;
    *(int4*)&vL[d*VSTR + kb8] = *(const int4*)&vT[(((size_t)i*NH + h)*DH + d)*NSEQ + kb8];
  }
  __syncthreads();

  // q fragment for this wave's 16 rows
  FragBF qf;
  {
    const bf16_t* p = qb + (size_t)(i*NSEQ + j0 + m)*CDIM + h*DH + hf*8;
    qf.q[0] = *(const int4*)p;
    qf.q[1] = *(const int4*)(p + 16);
  }
  float pmrow[8];
#pragma unroll
  for (int r = 0; r < 8; ++r) pmrow[r] = pm[i*NSEQ + j0 + r + 8*hf];

  float m_run[8], l_run[8];
#pragma unroll
  for (int r = 0; r < 8; ++r) { m_run[r] = -1e30f; l_run[r] = 0.0f; }
  v8f o0 = {0.f,0.f,0.f,0.f,0.f,0.f,0.f,0.f};
  v8f o1 = {0.f,0.f,0.f,0.f,0.f,0.f,0.f,0.f};

  const float scale = 0.17677669529663687f;  // 1/sqrt(32)
  bf16_t* myP = &pL[wave*16*KSTR];

#pragma unroll 1
  for (int kc = 0; kc < NSEQ; kc += 32) {
    // S = q . k^T for two 16-column tiles (preload both B fragments)
    FragBF b0, b1;
    {
      const bf16_t* p0 = kL + (kc + m)*KSTR + hf*8;
      b0.q[0] = *(const int4*)p0;       b0.q[1] = *(const int4*)(p0 + 16);
      const bf16_t* p1 = kL + (kc + 16 + m)*KSTR + hf*8;
      b1.q[0] = *(const int4*)p1;       b1.q[1] = *(const int4*)(p1 + 16);
    }
    v8f s0 = {0.f,0.f,0.f,0.f,0.f,0.f,0.f,0.f};
    v8f s1 = {0.f,0.f,0.f,0.f,0.f,0.f,0.f,0.f};
    s0 = __builtin_amdgcn_wmma_f32_16x16x32_bf16(false, qf.v, false, b0.v, (short)0, s0, false, false);
    s1 = __builtin_amdgcn_wmma_f32_16x16x32_bf16(false, qf.v, false, b1.v, (short)0, s1, false, false);

    const float pmk0 = pm[i*NSEQ + kc + m];
    const float pmk1 = pm[i*NSEQ + kc + 16 + m];
#pragma unroll
    for (int r = 0; r < 8; ++r) {
      const int jrow = j0 + r + 8*hf;
      const float* bp = biasT + (size_t)h*NROWS + (size_t)jrow*NSEQ + kc + m;
      float x0 = s0[r]*scale + bp[0];
      float x1 = s1[r]*scale + bp[16];
      if (pmrow[r]*pmk0 <= 0.5f) x0 = -1e9f;
      if (pmrow[r]*pmk1 <= 0.5f) x1 = -1e9f;
      s0[r] = x0; s1[r] = x1;
    }
    // online softmax update (stats replicated across each 16-lane half = D layout)
#pragma unroll
    for (int r = 0; r < 8; ++r) {
      float cmax = redmax16(fmaxf(s0[r], s1[r]));
      const float mnew = fmaxf(m_run[r], cmax);
      const float alpha = __expf(m_run[r] - mnew);
      m_run[r] = mnew;
      l_run[r] *= alpha;
      o0[r] *= alpha; o1[r] *= alpha;
      const float p0 = __expf(s0[r] - mnew);
      const float p1 = __expf(s1[r] - mnew);
      s0[r] = p0; s1[r] = p1;
      l_run[r] += redsum16(p0 + p1);
    }
    // D-layout -> A-fragment layout via per-wave LDS staging (same-wave DS ops are in-order)
#pragma unroll
    for (int r = 0; r < 8; ++r) {
      const int pr = r + 8*hf;
      myP[pr*KSTR + m]      = (bf16_t)s0[r];
      myP[pr*KSTR + 16 + m] = (bf16_t)s1[r];
    }
    FragBF pf, v0f, v1f;
    {
      const bf16_t* p = myP + m*KSTR + hf*8;
      pf.q[0] = *(const int4*)p; pf.q[1] = *(const int4*)(p + 16);
      const bf16_t* p0 = vL + m*VSTR + kc + hf*8;
      v0f.q[0] = *(const int4*)p0;  v0f.q[1] = *(const int4*)(p0 + 16);
      const bf16_t* p1 = vL + (16 + m)*VSTR + kc + hf*8;
      v1f.q[0] = *(const int4*)p1;  v1f.q[1] = *(const int4*)(p1 + 16);
    }
    o0 = __builtin_amdgcn_wmma_f32_16x16x32_bf16(false, pf.v, false, v0f.v, (short)0, o0, false, false);
    o1 = __builtin_amdgcn_wmma_f32_16x16x32_bf16(false, pf.v, false, v1f.v, (short)0, o1, false, false);
  }

  // normalize, gate, store
#pragma unroll
  for (int r = 0; r < 8; ++r) {
    const float inv = (l_run[r] > 0.0f) ? fast_rcp(l_run[r]) : 0.0f;
    const int jrow = j0 + r + 8*hf;
    const size_t base = (size_t)(i*NSEQ + jrow)*CDIM + h*DH;
    const float g0 = (float)gb[base + m];
    const float g1 = (float)gb[base + 16 + m];
    ob[base + m]      = (bf16_t)(o0[r]*inv*g0);
    ob[base + 16 + m] = (bf16_t)(o1[r]*inv*g1);
  }
}

// ---------------- Kernel 4: output projection ----------------
__global__ __launch_bounds__(256)
void out_kernel(const bf16_t* __restrict__ ob, const float* __restrict__ Wo,
                const float* __restrict__ bo, float* __restrict__ out)
{
  __shared__ __align__(16) bf16_t wL[CDIM*WSTR];
  __shared__ float boL[CDIM];

  const int tid  = threadIdx.x;
  const int wave = tid >> 5;
  const int lane = tid & 31;
  const int m  = lane & 15;
  const int hf = lane >> 4;
  const int row0 = blockIdx.x * 128 + wave*16;

  for (int idx = tid; idx < CDIM; idx += 256) boL[idx] = bo[idx];
  for (int idx = tid; idx < CDIM*CDIM; idx += 256) {
    const int n = idx >> 7, k = idx & 127;
    wL[n*WSTR + k] = (bf16_t)Wo[idx];
  }
  __syncthreads();

  FragBF a[4];
  {
    const bf16_t* p = ob + (size_t)(row0 + m)*CDIM + hf*8;
#pragma unroll
    for (int kk=0; kk<4; ++kk) {
      a[kk].q[0] = *(const int4*)(p + kk*32);
      a[kk].q[1] = *(const int4*)(p + kk*32 + 16);
    }
  }

#pragma unroll 1
  for (int nt = 0; nt < 8; ++nt) {
    FragBF bfr[4];
#pragma unroll
    for (int kk = 0; kk < 4; ++kk) {
      const bf16_t* p = wL + (nt*16 + m)*WSTR + kk*32 + hf*8;
      bfr[kk].q[0] = *(const int4*)p;
      bfr[kk].q[1] = *(const int4*)(p + 16);
    }
    v8f acc = {0.f,0.f,0.f,0.f,0.f,0.f,0.f,0.f};
#pragma unroll
    for (int kk = 0; kk < 4; ++kk)
      acc = __builtin_amdgcn_wmma_f32_16x16x32_bf16(false, a[kk].v, false, bfr[kk].v,
                                                    (short)0, acc, false, false);
    const int col = nt*16 + m;
    const float bb = boL[col];
#pragma unroll
    for (int r = 0; r < 8; ++r)
      out[(size_t)(row0 + r + 8*hf)*CDIM + col] = acc[r] + bb;
  }
}

extern "C" void kernel_launch(void* const* d_in, const int* in_sizes, int n_in,
                              void* d_out, int out_size, void* d_ws, size_t ws_size,
                              hipStream_t stream) {
  (void)in_sizes; (void)n_in; (void)out_size; (void)ws_size;
  const float* z   = (const float*)d_in[0];
  const float* pmv = (const float*)d_in[1];
  const float* lnw = (const float*)d_in[2];
  const float* lnb = (const float*)d_in[3];
  const float* Wq  = (const float*)d_in[4];
  const float* Wk  = (const float*)d_in[5];
  const float* Wv  = (const float*)d_in[6];
  const float* Wb  = (const float*)d_in[7];
  const float* Wg  = (const float*)d_in[8];
  const float* bg  = (const float*)d_in[9];
  const float* Wo  = (const float*)d_in[10];
  const float* bo  = (const float*)d_in[11];
  float* out = (float*)d_out;

  const size_t SZ = (size_t)NROWS * CDIM * sizeof(bf16_t);   // 16 MB per bf16 tensor
  char* ws = (char*)d_ws;
  bf16_t* zn    = (bf16_t*)(ws);            // reused as ob after projections
  bf16_t* qb    = (bf16_t*)(ws + SZ);
  bf16_t* kb    = (bf16_t*)(ws + 2*SZ);
  bf16_t* vT    = (bf16_t*)(ws + 3*SZ);
  bf16_t* gb    = (bf16_t*)(ws + 4*SZ);
  float*  biasT = (float*) (ws + 5*SZ);     // 1 MB
  bf16_t* ob    = zn;                       // zn dead after proj_kernel

  ln_kernel  <<<NROWS/8,   256, 0, stream>>>(z, lnw, lnb, zn);
  proj_kernel<<<NROWS/128, 256, 0, stream>>>(zn, Wq, Wk, Wv, Wg, Wb, bg,
                                             qb, kb, vT, gb, biasT);
  attn_kernel<<<NSEQ*NH*4, 128, 0, stream>>>(qb, kb, vT, gb, biasT, pmv, ob);
  out_kernel <<<NROWS/128, 256, 0, stream>>>(ob, Wo, bo, out);
}
